// MultiHeadGATMethod_86388972192205
// MI455X (gfx1250) — matrix operand order repaired and checked by reading, not compiled, not run on hardware
//
#include <hip/hip_runtime.h>
#include <hip/hip_bf16.h>
#include <cstdint>
#include <cstddef>

typedef __attribute__((ext_vector_type(16))) __bf16 v16bf;
typedef __attribute__((ext_vector_type(8)))  __bf16 v8bf;
typedef __attribute__((ext_vector_type(8)))  float  v8f;

#define IN_DIM    128
#define OUT_DIM   64
#define N_HEADS   4
#define NEG_SLOPE 0.2f
#define GAT_EPS   1e-8f

// f32 -> bf16 bits, round-to-nearest-even
static __device__ __forceinline__ unsigned f2bfbits(float f) {
    unsigned u = __builtin_bit_cast(unsigned, f);
    u += 0x7FFFu + ((u >> 16) & 1u);
    return u >> 16;
}

// float atomic-max via sign-split int/uint atomics (no CAS loop)
static __device__ __forceinline__ void atomicMaxFloatBits(int* addr, float val) {
    if (!(val < 0.0f)) {
        atomicMax(addr, __float_as_int(val));
    } else {
        atomicMin(reinterpret_cast<unsigned int*>(addr), __float_as_uint(val));
    }
}

// ---------------------------------------------------------------------------
// Stage 0: one-time f32 -> bf16 conversion (4 elems/thread, packed stores)
// ---------------------------------------------------------------------------
__global__ void __launch_bounds__(256)
conv_f32_bf16(const float4* __restrict__ src, uint2* __restrict__ dst, int n4)
{
    const int i = blockIdx.x * blockDim.x + threadIdx.x;
    if (i >= n4) return;
    const float4 f = src[i];
    uint2 o;
    o.x = f2bfbits(f.x) | (f2bfbits(f.y) << 16);
    o.y = f2bfbits(f.z) | (f2bfbits(f.w) << 16);
    dst[i] = o;
}

// ---------------------------------------------------------------------------
// Stage 1: per-head fused GEMM  Q = lrelu(xWq^T), K = lrelu(xWk^T), H = xWh^T+b
// One workgroup (4 waves) per 16-node tile; wave w owns output cols [16w,16w+16)
// All operands pre-converted bf16; A fragments reused across Q/K/H.
// ---------------------------------------------------------------------------
__global__ void __launch_bounds__(128)
gat_qkh_gemm(const __bf16* __restrict__ xbf,   // [N,128] bf16
             const __bf16* __restrict__ Wq,    // [64,128] bf16 (this head)
             const __bf16* __restrict__ Wk,
             const __bf16* __restrict__ Wh,
             const float*  __restrict__ bh,    // [64] f32
             float* __restrict__ Qo,           // [N,64]
             float* __restrict__ Ko,
             float* __restrict__ Ho,
             int n_nodes)
{
    __shared__ __bf16 ldsA[16 * IN_DIM];   // 4 KB x-tile (bf16)

    const int tid   = threadIdx.x;
    const int wave  = tid >> 5;            // 0..3 -> N-tile of output
    const int lane  = tid & 31;
    const int half  = lane >> 4;           // 0/1
    const int lm    = lane & 15;
    const int tile0 = blockIdx.x * 16;
    const bool fullTile = (tile0 + 16 <= n_nodes);   // block-uniform

    // stage 16x128 bf16 tile -> LDS (straight b128 copies, no conversion)
    {
        const int r    = tid >> 3;         // 0..15
        const int c    = (tid & 7) * 16;   // 0..112
        const int node = tile0 + r;
        uint4 d0 = make_uint4(0u, 0u, 0u, 0u);
        uint4 d1 = make_uint4(0u, 0u, 0u, 0u);
        if (node < n_nodes) {
            const uint4* src =
                reinterpret_cast<const uint4*>(xbf + (size_t)node * IN_DIM + c);
            d0 = src[0];
            d1 = src[1];
        }
        uint4* dst = reinterpret_cast<uint4*>(&ldsA[r * IN_DIM + c]);
        dst[0] = d0;
        dst[1] = d1;
    }
    __syncthreads();

    // A fragments (16x32, contiguous k order matching B), loaded ONCE, reused for Q/K/H:
    // lane holds row lm; K = kc + half*16 + [0..15]
    v16bf afrag[4];
    #pragma unroll
    for (int kci = 0; kci < 4; ++kci) {
        const int kc = kci * 32;
        v8bf alo = *reinterpret_cast<const v8bf*>(
            &ldsA[lm * IN_DIM + kc + half * 16]);
        v8bf ahi = *reinterpret_cast<const v8bf*>(
            &ldsA[lm * IN_DIM + kc + half * 16 + 8]);
        afrag[kci] = __builtin_shufflevector(alo, ahi,
            0,1,2,3,4,5,6,7,8,9,10,11,12,13,14,15);
    }

    const int ncol = wave * 16 + lm;       // output column owned by this lane

    const __bf16* Ws[3] = { Wq, Wk, Wh };
    float*        Os[3] = { Qo, Ko, Ho };

    #pragma unroll
    for (int mat = 0; mat < 3; ++mat) {
        const __bf16* __restrict__ W = Ws[mat];
        v8f acc = {};
        #pragma unroll
        for (int kci = 0; kci < 4; ++kci) {
            // B frag: B[k][n] = W[n][k]; lane holds col ncol,
            // K = kc + half*16 + [0..15] -> one 32B contiguous bf16 load
            v16bf b = *reinterpret_cast<const v16bf*>(
                W + (size_t)ncol * IN_DIM + kci * 32 + half * 16);
            acc = __builtin_amdgcn_wmma_f32_16x16x32_bf16(
                false, afrag[kci], false, b, (short)0, acc, false, false);
        }

        // epilogue: C/D layout -> lane owns rows v + 8*half, col ncol
        const float bias = (mat == 2) ? bh[ncol] : 0.f;
        float* __restrict__ O = Os[mat];
        // single base address; row v at immediate offset v*OUT_DIM*4 bytes
        float* Obase = O + ((size_t)(tile0 + 8 * half) * OUT_DIM + ncol);
        if (fullTile) {                     // uniform fast path: no exec churn
            #pragma unroll
            for (int v = 0; v < 8; ++v) {
                float val = acc[v];
                if (mat < 2) val = (val > 0.f) ? val : NEG_SLOPE * val;
                else         val += bias;
                Obase[v * OUT_DIM] = val;
            }
        } else {
            #pragma unroll
            for (int v = 0; v < 8; ++v) {
                if (tile0 + v + 8 * half < n_nodes) {
                    float val = acc[v];
                    if (mat < 2) val = (val > 0.f) ? val : NEG_SLOPE * val;
                    else         val += bias;
                    Obase[v * OUT_DIM] = val;
                }
            }
        }
    }
}

// ---------------------------------------------------------------------------
// Stage 2: edge scores + segment max.  One wave per edge (2 cols per lane).
// ---------------------------------------------------------------------------
__global__ void __launch_bounds__(256)
gat_edge_scores(const float* __restrict__ Q, const float* __restrict__ K,
                const long long* __restrict__ erow,
                const long long* __restrict__ ecol,
                float* __restrict__ scores, int* __restrict__ mbits,
                int n_edges)
{
    const int lane = threadIdx.x & 31;
    const int e    = blockIdx.x * 8 + (threadIdx.x >> 5);
    if (e >= n_edges) return;
    const int r = (int)erow[e];
    const int c = (int)ecol[e];
    const float2 q = *reinterpret_cast<const float2*>(
        Q + (size_t)r * OUT_DIM + lane * 2);
    const float2 k = *reinterpret_cast<const float2*>(
        K + (size_t)c * OUT_DIM + lane * 2);
    float p = q.x * k.x + q.y * k.y;
    #pragma unroll
    for (int off = 16; off > 0; off >>= 1)
        p += __shfl_xor(p, off, 32);
    if (lane == 0) {
        const float s = p * 0.125f;        // / sqrt(OUT_DIM)
        scores[e] = s;
        atomicMaxFloatBits(mbits + r, s);
    }
}

// ---------------------------------------------------------------------------
// Stage 3: e = exp(s - m[row]); z[row] += e
// ---------------------------------------------------------------------------
__global__ void __launch_bounds__(256)
gat_edge_exp(float* __restrict__ scores,
             const long long* __restrict__ erow,
             const int* __restrict__ mbits,
             float* __restrict__ z, int n_edges)
{
    const int e = blockIdx.x * blockDim.x + threadIdx.x;
    if (e >= n_edges) return;
    const int r  = (int)erow[e];
    const float m  = __int_as_float(mbits[r]);
    const float ev = __expf(scores[e] - m);
    scores[e] = ev;
    atomicAdd(z + r, ev);
}

// ---------------------------------------------------------------------------
// Stage 4: out[row] += (alpha/4) * H[col].  One wave per edge.
// ---------------------------------------------------------------------------
__global__ void __launch_bounds__(256)
gat_edge_scatter(const float* __restrict__ scores,
                 const long long* __restrict__ erow,
                 const long long* __restrict__ ecol,
                 const float* __restrict__ z,
                 const float* __restrict__ H,
                 float* __restrict__ out, int n_edges)
{
    const int lane = threadIdx.x & 31;
    const int e    = blockIdx.x * 8 + (threadIdx.x >> 5);
    if (e >= n_edges) return;
    const int r = (int)erow[e];
    const int c = (int)ecol[e];
    float a = 0.f;
    if (lane == 0) a = 0.25f * scores[e] / (z[r] + GAT_EPS);
    a = __shfl(a, 0, 32);
    const float2 h = *reinterpret_cast<const float2*>(
        H + (size_t)c * OUT_DIM + lane * 2);
    float* o = out + (size_t)r * OUT_DIM + lane * 2;
    atomicAdd(o + 0, a * h.x);
    atomicAdd(o + 1, a * h.y);
}

// ---------------------------------------------------------------------------
__global__ void gat_zero(float* __restrict__ p, int n) {
    const int i = blockIdx.x * blockDim.x + threadIdx.x;
    if (i < n) p[i] = 0.f;
}
__global__ void gat_init_mz(int* __restrict__ mbits, float* __restrict__ z, int n) {
    const int i = blockIdx.x * blockDim.x + threadIdx.x;
    if (i < n) { mbits[i] = (int)0xFF800000u; z[i] = 0.f; }   // -inf bits
}

// ---------------------------------------------------------------------------
extern "C" void kernel_launch(void* const* d_in, const int* in_sizes, int n_in,
                              void* d_out, int out_size, void* d_ws, size_t ws_size,
                              hipStream_t stream)
{
    const float*     x   = (const float*)d_in[0];
    const long long* ei  = (const long long*)d_in[1];   // int64 [2, E]
    const float*     Wq  = (const float*)d_in[2];       // [4,64,128]
    const float*     Wk  = (const float*)d_in[3];
    const float*     Wh  = (const float*)d_in[4];
    const float*     bh  = (const float*)d_in[5];       // [4,64]

    const int N = in_sizes[0] / IN_DIM;
    const int E = in_sizes[1] / 2;
    const long long* erow = ei;
    const long long* ecol = ei + E;

    const int WELEMS = N_HEADS * OUT_DIM * IN_DIM;      // per weight tensor

    // workspace carve-out (~110 MB)
    char* ws = (char*)d_ws;
    size_t off = 0;
    auto carve = [&](size_t bytes) -> void* {
        void* p = ws + off;
        off += (bytes + 255) & ~(size_t)255;
        return p;
    };
    float*  Qb     = (float*)carve((size_t)N * OUT_DIM * sizeof(float));
    float*  Kb     = (float*)carve((size_t)N * OUT_DIM * sizeof(float));
    float*  Hb     = (float*)carve((size_t)N * OUT_DIM * sizeof(float));
    float*  scores = (float*)carve((size_t)E * sizeof(float));
    int*    mbits  = (int*)  carve((size_t)N * sizeof(int));
    float*  zb     = (float*)carve((size_t)N * sizeof(float));
    __bf16* xbf    = (__bf16*)carve((size_t)N * IN_DIM * sizeof(__bf16));
    __bf16* Wqbf   = (__bf16*)carve((size_t)WELEMS * sizeof(__bf16));
    __bf16* Wkbf   = (__bf16*)carve((size_t)WELEMS * sizeof(__bf16));
    __bf16* Whbf   = (__bf16*)carve((size_t)WELEMS * sizeof(__bf16));

    float* out = (float*)d_out;

    // one-time (per launch) bf16 pre-conversion
    {
        const int nx4 = (N * IN_DIM) / 4;
        conv_f32_bf16<<<(nx4 + 255) / 256, 256, 0, stream>>>(
            (const float4*)x, (uint2*)xbf, nx4);
        const int nw4 = WELEMS / 4;
        conv_f32_bf16<<<(nw4 + 255) / 256, 256, 0, stream>>>(
            (const float4*)Wq, (uint2*)Wqbf, nw4);
        conv_f32_bf16<<<(nw4 + 255) / 256, 256, 0, stream>>>(
            (const float4*)Wk, (uint2*)Wkbf, nw4);
        conv_f32_bf16<<<(nw4 + 255) / 256, 256, 0, stream>>>(
            (const float4*)Wh, (uint2*)Whbf, nw4);
    }

    const int outElems = N * OUT_DIM;
    gat_zero<<<(outElems + 255) / 256, 256, 0, stream>>>(out, outElems);

    const int tiles   = (N + 15) / 16;
    const int eblocks = (E + 7) / 8;       // 8 waves (edges) per 256-thread block

    for (int h = 0; h < N_HEADS; ++h) {
        const size_t woff = (size_t)h * OUT_DIM * IN_DIM;
        gat_init_mz<<<(N + 255) / 256, 256, 0, stream>>>(mbits, zb, N);
        gat_qkh_gemm<<<tiles, 128, 0, stream>>>(
            xbf, Wqbf + woff, Wkbf + woff, Whbf + woff,
            bh + (size_t)h * OUT_DIM, Qb, Kb, Hb, N);
        gat_edge_scores<<<eblocks, 256, 0, stream>>>(
            Qb, Kb, erow, ecol, scores, mbits, E);
        gat_edge_exp<<<(E + 255) / 256, 256, 0, stream>>>(
            scores, erow, mbits, zb, E);
        gat_edge_scatter<<<eblocks, 256, 0, stream>>>(
            scores, erow, ecol, zb, Hb, out, E);
    }
}
